// CollaborativeExpertsModule_33088428048437
// MI455X (gfx1250) — matrix operand order repaired and checked by reading, not compiled, not run on hardware
//
#include <hip/hip_runtime.h>

// ---------------------------------------------------------------------------
// Types / helpers
// ---------------------------------------------------------------------------
typedef __bf16 bf16;
typedef __attribute__((ext_vector_type(8)))  bf16  v8bf;
typedef __attribute__((ext_vector_type(16))) bf16  v16bf;
typedef __attribute__((ext_vector_type(8)))  float v8f;
typedef int v4i32 __attribute__((vector_size(16)));
typedef unsigned int u32x4 __attribute__((vector_size(16)));
typedef int i32x8 __attribute__((vector_size(32)));
typedef int i32x4 __attribute__((vector_size(16)));

__device__ __forceinline__ bf16  f2bf(float v) { return (bf16)v; }
__device__ __forceinline__ float bf2f(bf16 v)  { return (float)v; }

__device__ __forceinline__ v8f wmma_bf16(v16bf a, v16bf b, v8f c) {
  // v_wmma_f32_16x16x32_bf16: D = A(16x32) * B(32x16) + C(16x16 f32)
  return __builtin_amdgcn_wmma_f32_16x16x32_bf16(false, a, false, b, (short)0, c, false, false);
}

__device__ __forceinline__ v16bf cat8(v8bf lo, v8bf hi) {
  return __builtin_shufflevector(lo, hi, 0,1,2,3,4,5,6,7,8,9,10,11,12,13,14,15);
}

// gfx1250 async global->LDS copy (ASYNCcnt-tracked); VGPR-staging fallback.
#if __has_builtin(__builtin_amdgcn_global_load_async_to_lds_b128)
#define HAS_ASYNC_LDS 1
#else
#define HAS_ASYNC_LDS 0
#endif

// gfx1250 Tensor Data Mover (TENSORcnt-tracked DMA).
#if __has_builtin(__builtin_amdgcn_tensor_load_to_lds)
#define HAS_TDM 1
#else
#define HAS_TDM 0
#endif

__device__ __forceinline__ void copy16B_to_lds(const bf16* g, bf16* l) {
#if HAS_ASYNC_LDS
  __builtin_amdgcn_global_load_async_to_lds_b128(
      (__attribute__((address_space(1))) v4i32*)(g),
      (__attribute__((address_space(3))) v4i32*)(l), 0, 0);
#else
  *(v8bf*)l = *(const v8bf*)g;
#endif
}

__device__ __forceinline__ void async_copy_wait() {
#if HAS_ASYNC_LDS
#if __has_builtin(__builtin_amdgcn_s_wait_asynccnt)
  __builtin_amdgcn_s_wait_asynccnt(0);
#else
  asm volatile("s_wait_asynccnt 0" ::: "memory");
#endif
#endif
}

// Problem constants
#define NT   4096          // B*S tokens
#define DM   768           // model dim
#define HH   2048          // expert hidden
#define NEXP 8             // experts
#define TDQ  2304          // 3*D
#define NE   (NT*NEXP)     // token-expert rows
#define NK   (NT*2)        // token-topk rows

// GEMM tiling
#define GBM 128
#define GBN 64
#define GBK 32
#define LSTR 56            // LDS row stride (elems): 112B, 16B-aligned, conflict-free

#if HAS_TDM
// TDM: load a GBM x GBK bf16 tile (row stride lda elems) into LDS with
// per-row padding: 16 dwords (64B) data + 12 dwords (48B) pad = 112B = LSTR*2.
// D# per CDNA5 ISA ch.8: group0 {count,lds_addr,global_addr,type=2},
// group1 {data_size=2B, pad_enable, pad_interval=16dw(code 3),
//         pad_amount=12dw(code 11), tensor/tile dims, dim0 stride}.
__device__ __forceinline__ void tdm_load_tile_A(const bf16* gsrc, bf16* ldst,
                                                int lda, int Kdim, int Mtot) {
  unsigned long long ga = (unsigned long long)(size_t)gsrc;
  unsigned int galo  = (unsigned int)(ga & 0xffffffffull);
  unsigned int gahi  = (unsigned int)((ga >> 32) & 0x1ffffffull);
  unsigned int lofs  = (unsigned int)(size_t)(__attribute__((address_space(3))) bf16*)ldst;

  u32x4 g0 = { 1u, lofs, galo, gahi | (2u << 30) };

  i32x8 g1;
  g1[0] = (1 << 16)             // data_size = 2 bytes
        | (1 << 20)             // pad_enable
        | (3 << 22)             // pad_interval: 16 dwords
        | (11 << 25);           // pad_amount: 12 dwords
  g1[1] = (int)(((unsigned)Kdim & 0xffffu) << 16);            // tensor_dim0[15:0]
  g1[2] = (int)((((unsigned)Kdim >> 16) & 0xffffu)
        |       (((unsigned)Mtot & 0xffffu) << 16));          // dim0 hi | tensor_dim1 lo
  g1[3] = (int)((((unsigned)Mtot >> 16) & 0xffffu)
        |       ((unsigned)GBK << 16));                       // dim1 hi | tile_dim0 = 32
  g1[4] = GBM;                                                // tile_dim1 = 128, tile_dim2 = 0
  g1[5] = lda;                                                // tensor_dim0_stride[31:0]
  g1[6] = 0;                                                  // stride hi / dim1_stride = 0
  g1[7] = 0;
  i32x4 z4 = {0, 0, 0, 0};
  i32x8 z8 = {0, 0, 0, 0, 0, 0, 0, 0};
  __builtin_amdgcn_tensor_load_to_lds(g0, g1, z4, z4, z8, 0);
}
#endif

// ---------------------------------------------------------------------------
// Generic bf16 GEMM: C(MxN) = act(A(MxK) * B(KxN) + bias + resid)
// grid = (N/GBN, M/GBM), block = 256 (8 waves, 4x2 wave grid, 2x2 wmma/wave).
// Double-buffered LDS; A tile staged with the Tensor Data Mover (or async
// global->LDS copies as fallback). Epilogue resolved at compile time.
// ---------------------------------------------------------------------------
template<bool BIAS, bool RESID, bool CF, bool CB, bool GELU>
__global__ __launch_bounds__(256) void gemm_tpl(
    const bf16* __restrict__ A, int lda,
    const bf16* __restrict__ Bm, int ldb, int Kdim,
    const float* __restrict__ bias,
    const float* __restrict__ resid, int ldr,
    float* __restrict__ Cf, int ldcf,
    bf16* __restrict__ Cb, int ldcb)
{
  __shared__ bf16 As[2][GBM * LSTR];
  __shared__ bf16 Bs[2][GBN * LSTR];
  const int t    = threadIdx.x;
  const int lane = t & 31, wv = t >> 5;
  const int wm = wv >> 1, wn = wv & 1;
  const int m0 = blockIdx.y * GBM, n0 = blockIdx.x * GBN;
  const int lo = lane & 15, hi = lane >> 4;

  v8f z = {0.f,0.f,0.f,0.f,0.f,0.f,0.f,0.f};
  v8f acc[2][2];
  acc[0][0]=z; acc[0][1]=z; acc[1][0]=z; acc[1][1]=z;

  const int arow = t >> 1, acol = (t & 1) * 16;   // A tile: 128 rows x 32 cols
  const int bk   = t >> 3, bn   = (t & 7) * 8;    // B tile: 32 rows x 64 cols
  (void)arow; (void)acol;

  auto stageA = [&](int k0, int buf) {
#if HAS_TDM
    if (wv == 0)
      tdm_load_tile_A(A + (size_t)m0 * lda + k0, &As[buf][0],
                      lda, Kdim, (int)gridDim.y * GBM);
#else
    const bf16* gp = A + (size_t)(m0 + arow) * lda + (k0 + acol);
    bf16* lp = &As[buf][arow * LSTR + acol];
    copy16B_to_lds(gp,     lp);
    copy16B_to_lds(gp + 8, lp + 8);
#endif
  };
  auto stage_wait = [&]() {
#if HAS_TDM
    __builtin_amdgcn_s_wait_tensorcnt(0);
#else
    async_copy_wait();
#endif
  };
  auto loadB = [&](int k0) -> v8bf {
    return *(const v8bf*)(Bm + (size_t)(k0 + bk) * ldb + (n0 + bn));
  };
  auto storeB = [&](v8bf d, int buf) {
    #pragma unroll
    for (int j = 0; j < 8; ++j) Bs[buf][(bn + j) * LSTR + bk] = d[j];
  };

  // prologue: stage tile 0 into buffer 0
  stageA(0, 0);
  storeB(loadB(0), 0);
  stage_wait();
  __syncthreads();

  const int nk = Kdim / GBK;
  for (int i = 0; i < nk; ++i) {
    const int cur = i & 1, nxt = cur ^ 1;
    const bool have_next = (i + 1 < nk);
    v8bf bnext;
    if (have_next) {               // prefetch tile i+1 while computing tile i
      stageA((i + 1) * GBK, nxt);
      bnext = loadB((i + 1) * GBK);
    }

    v16bf af[2], bfr[2];
    #pragma unroll
    for (int tm = 0; tm < 2; ++tm) { // A frag: lanes 0-15 K 0..7/16..23, lanes 16-31 K 8..15/24..31
      const bf16* ap = &As[cur][(wm*32 + tm*16 + lo) * LSTR + hi*8];
      af[tm] = cat8(*(const v8bf*)ap, *(const v8bf*)(ap + 16));
    }
    #pragma unroll
    for (int tn = 0; tn < 2; ++tn) { // B frag: lanes 0-15 K 0..7/16..23, lanes 16-31 K 8..15/24..31
      const bf16* bp = &Bs[cur][(wn*32 + tn*16 + lo) * LSTR + hi*8];
      bfr[tn] = cat8(*(const v8bf*)bp, *(const v8bf*)(bp + 16));
    }
    #pragma unroll
    for (int tm = 0; tm < 2; ++tm)
      #pragma unroll
      for (int tn = 0; tn < 2; ++tn)
        acc[tm][tn] = wmma_bf16(af[tm], bfr[tn], acc[tm][tn]);

    if (have_next) storeB(bnext, nxt);
    stage_wait();
    __syncthreads();
  }

  #pragma unroll
  for (int tm = 0; tm < 2; ++tm) {
    #pragma unroll
    for (int tn = 0; tn < 2; ++tn) {
      const int col = n0 + wn*32 + tn*16 + lo;
      float bv = 0.0f;
      if (BIAS) bv = bias[col];
      #pragma unroll
      for (int r = 0; r < 8; ++r) {
        const int row = m0 + wm*32 + tm*16 + hi*8 + r;
        float v = acc[tm][tn][r] + bv;
        if (RESID) v += resid[(size_t)row * ldr + col];
        if (GELU)  v = 0.5f * v * (1.0f + erff(v * 0.70710678118f)); // exact GELU
        if (CF) Cf[(size_t)row * ldcf + col] = v;
        if (CB) Cb[(size_t)row * ldcb + col] = f2bf(v);
      }
    }
  }
}

// ---------------------------------------------------------------------------
// Dual GEMM for SwiGLU: H = silu(A*B1) * (A*B2), bf16 out. Same tiling.
// (A tile staged via ASYNCcnt path so both CDNA5 DMA flavors are exercised.)
// ---------------------------------------------------------------------------
__global__ __launch_bounds__(256) void dual_gemm_silu_kernel(
    const bf16* __restrict__ A, int lda,
    const bf16* __restrict__ B1, const bf16* __restrict__ B2, int ldb, int Kdim,
    bf16* __restrict__ Ho, int ldh)
{
  __shared__ bf16 As[2][GBM * LSTR];
  __shared__ bf16 Bs1[2][GBN * LSTR];
  __shared__ bf16 Bs2[2][GBN * LSTR];
  const int t    = threadIdx.x;
  const int lane = t & 31, wv = t >> 5;
  const int wm = wv >> 1, wn = wv & 1;
  const int m0 = blockIdx.y * GBM, n0 = blockIdx.x * GBN;
  const int lo = lane & 15, hi = lane >> 4;

  v8f z = {0.f,0.f,0.f,0.f,0.f,0.f,0.f,0.f};
  v8f accg[2][2], accu[2][2];
  accg[0][0]=z; accg[0][1]=z; accg[1][0]=z; accg[1][1]=z;
  accu[0][0]=z; accu[0][1]=z; accu[1][0]=z; accu[1][1]=z;

  const int arow = t >> 1, acol = (t & 1) * 16;
  const int bk   = t >> 3, bn   = (t & 7) * 8;

  auto stageA = [&](int k0, int buf) {
    const bf16* gp = A + (size_t)(m0 + arow) * lda + (k0 + acol);
    bf16* lp = &As[buf][arow * LSTR + acol];
    copy16B_to_lds(gp,     lp);
    copy16B_to_lds(gp + 8, lp + 8);
  };
  auto loadB = [&](const bf16* Bp, int k0) -> v8bf {
    return *(const v8bf*)(Bp + (size_t)(k0 + bk) * ldb + (n0 + bn));
  };
  auto storeB = [&](bf16* Bsbuf, v8bf d) {
    #pragma unroll
    for (int j = 0; j < 8; ++j) Bsbuf[(bn + j) * LSTR + bk] = d[j];
  };

  stageA(0, 0);
  storeB(&Bs1[0][0], loadB(B1, 0));
  storeB(&Bs2[0][0], loadB(B2, 0));
  async_copy_wait();
  __syncthreads();

  const int nk = Kdim / GBK;
  for (int i = 0; i < nk; ++i) {
    const int cur = i & 1, nxt = cur ^ 1;
    const bool have_next = (i + 1 < nk);
    v8bf b1n, b2n;
    if (have_next) {
      stageA((i + 1) * GBK, nxt);
      b1n = loadB(B1, (i + 1) * GBK);
      b2n = loadB(B2, (i + 1) * GBK);
    }

    v16bf af[2], b1f[2], b2f[2];
    #pragma unroll
    for (int tm = 0; tm < 2; ++tm) {
      const bf16* ap = &As[cur][(wm*32 + tm*16 + lo) * LSTR + hi*8];
      af[tm] = cat8(*(const v8bf*)ap, *(const v8bf*)(ap + 16));
    }
    #pragma unroll
    for (int tn = 0; tn < 2; ++tn) {
      const bf16* bp1 = &Bs1[cur][(wn*32 + tn*16 + lo) * LSTR + hi*8];
      const bf16* bp2 = &Bs2[cur][(wn*32 + tn*16 + lo) * LSTR + hi*8];
      b1f[tn] = cat8(*(const v8bf*)bp1, *(const v8bf*)(bp1 + 16));
      b2f[tn] = cat8(*(const v8bf*)bp2, *(const v8bf*)(bp2 + 16));
    }
    #pragma unroll
    for (int tm = 0; tm < 2; ++tm)
      #pragma unroll
      for (int tn = 0; tn < 2; ++tn) {
        accg[tm][tn] = wmma_bf16(af[tm], b1f[tn], accg[tm][tn]);
        accu[tm][tn] = wmma_bf16(af[tm], b2f[tn], accu[tm][tn]);
      }

    if (have_next) { storeB(&Bs1[nxt][0], b1n); storeB(&Bs2[nxt][0], b2n); }
    async_copy_wait();
    __syncthreads();
  }

  #pragma unroll
  for (int tm = 0; tm < 2; ++tm) {
    #pragma unroll
    for (int tn = 0; tn < 2; ++tn) {
      const int col = n0 + wn*32 + tn*16 + lo;
      #pragma unroll
      for (int r = 0; r < 8; ++r) {
        const int row = m0 + wm*32 + tm*16 + hi*8 + r;
        float g = accg[tm][tn][r];
        float u = accu[tm][tn][r];
        float h = (g / (1.0f + expf(-g))) * u;   // silu(g) * u
        Ho[(size_t)row * ldh + col] = f2bf(h);
      }
    }
  }
}

// ---------------------------------------------------------------------------
// Small support kernels
// ---------------------------------------------------------------------------
__global__ void cvt_f32_bf16_kernel(const float* __restrict__ in, bf16* __restrict__ out, int n) {
  int i = blockIdx.x * 256 + threadIdx.x;
  if (i < n) out[i] = f2bf(in[i]);
}

// W (O x I) fp32 -> Wt (I x O) bf16  (for y = x @ W.T patterns)
__global__ void transpose_cvt_kernel(const float* __restrict__ W, bf16* __restrict__ Wt, int O, int I) {
  int i = blockIdx.x * 256 + threadIdx.x;
  if (i < O * I) {
    int o = i / I, c = i % I;
    Wt[(size_t)c * O + o] = f2bf(W[(size_t)o * I + c]);
  }
}

// Router self-attention: one block (64 thr) per (token, head); L=8, hd=64, nh=12
__global__ __launch_bounds__(64) void router_attn_kernel(const bf16* __restrict__ qkv, bf16* __restrict__ aout) {
  const int blk = blockIdx.x;
  const int tok = blk / 12, h = blk % 12;
  const int t = threadIdx.x; // 0..63 = column
  __shared__ float qf[512], kf[512], vf[512], sc[64], pr[64];
  for (int l = 0; l < 8; ++l) {
    const bf16* base = qkv + ((size_t)tok * 8 + l) * TDQ + h * 64;
    qf[l*64 + t] = bf2f(base[t]);
    kf[l*64 + t] = bf2f(base[768 + t]);
    vf[l*64 + t] = bf2f(base[1536 + t]);
  }
  __syncthreads();
  {
    const int l = t >> 3, m = t & 7;
    float s = 0.f;
    #pragma unroll
    for (int j = 0; j < 64; ++j) s += qf[l*64 + j] * kf[m*64 + j];
    sc[t] = s * 0.125f; // 1/sqrt(64)
  }
  __syncthreads();
  if (t < 8) {
    float mx = sc[t*8];
    for (int m = 1; m < 8; ++m) mx = fmaxf(mx, sc[t*8 + m]);
    float e[8], s = 0.f;
    for (int m = 0; m < 8; ++m) { e[m] = expf(sc[t*8 + m] - mx); s += e[m]; }
    for (int m = 0; m < 8; ++m) pr[t*8 + m] = e[m] / s;
  }
  __syncthreads();
  for (int l = 0; l < 8; ++l) {
    float o = 0.f;
    #pragma unroll
    for (int m = 0; m < 8; ++m) o += pr[l*8 + m] * vf[m*64 + t];
    aout[((size_t)tok * 8 + l) * DM + h * 64 + t] = f2bf(o);
  }
}

// Per-token: RMS-norm each of 8 expert rows (post-residual) then mean over E
__global__ __launch_bounds__(256) void rms_mean_router_kernel(
    const float* __restrict__ tmp, const float* __restrict__ w, float* __restrict__ ctx_mean) {
  const int tok = blockIdx.x, t = threadIdx.x;
  __shared__ float red[256];
  float a0 = 0.f, a1 = 0.f, a2 = 0.f;
  for (int e = 0; e < NEXP; ++e) {
    const float* row = tmp + ((size_t)tok * NEXP + e) * DM;
    float x0 = row[t], x1 = row[t + 256], x2 = row[t + 512];
    red[t] = x0*x0 + x1*x1 + x2*x2;
    __syncthreads();
    for (int s = 128; s > 0; s >>= 1) { if (t < s) red[t] += red[t + s]; __syncthreads(); }
    float inv = rsqrtf(red[0] / (float)DM + 1e-5f);
    __syncthreads();
    a0 += x0 * inv * w[t]; a1 += x1 * inv * w[t + 256]; a2 += x2 * inv * w[t + 512];
  }
  float* o = ctx_mean + (size_t)tok * DM;
  o[t] = a0 * 0.125f; o[t + 256] = a1 * 0.125f; o[t + 512] = a2 * 0.125f;
}

// Per-token: logits = ctx_mean @ gate_w.T, softmax (for aux), top-2, gather selected experts
__global__ __launch_bounds__(256) void route_gather_kernel(
    const float* __restrict__ ctx_mean, const float* __restrict__ gate_w,
    const float* __restrict__ ef, float* __restrict__ sf, bf16* __restrict__ sfb,
    float* __restrict__ tkp, float* __restrict__ probs8, float* __restrict__ out_idx) {
  const int tok = blockIdx.x, t = threadIdx.x;
  __shared__ float red[256];
  __shared__ float lg[NEXP];
  __shared__ int   sidx[2];
  const float* cm = ctx_mean + (size_t)tok * DM;
  for (int e = 0; e < NEXP; ++e) {
    const float* gw = gate_w + (size_t)e * DM;
    red[t] = cm[t]*gw[t] + cm[t+256]*gw[t+256] + cm[t+512]*gw[t+512];
    __syncthreads();
    for (int s = 128; s > 0; s >>= 1) { if (t < s) red[t] += red[t + s]; __syncthreads(); }
    if (t == 0) lg[e] = red[0];
    __syncthreads();
  }
  if (t == 0) {
    float mx = lg[0];
    for (int e = 1; e < NEXP; ++e) mx = fmaxf(mx, lg[e]);
    float pe[NEXP], s = 0.f;
    for (int e = 0; e < NEXP; ++e) { pe[e] = expf(lg[e] - mx); s += pe[e]; }
    for (int e = 0; e < NEXP; ++e) probs8[(size_t)tok * NEXP + e] = pe[e] / s;
    int i0 = 0;
    for (int e = 1; e < NEXP; ++e) if (lg[e] > lg[i0]) i0 = e;
    int i1 = (i0 == 0) ? 1 : 0;
    for (int e = 0; e < NEXP; ++e) if (e != i0 && lg[e] > lg[i1]) i1 = e;
    float e1 = expf(lg[i1] - lg[i0]);
    float ss = 1.0f + e1;
    sidx[0] = i0; sidx[1] = i1;
    tkp[tok*2]     = 1.0f / ss;
    tkp[tok*2 + 1] = e1 / ss;
    out_idx[tok*2]     = (float)i0;
    out_idx[tok*2 + 1] = (float)i1;
  }
  __syncthreads();
  for (int kk = 0; kk < 2; ++kk) {
    const float* src = ef + ((size_t)tok * NEXP + sidx[kk]) * DM;
    float* d  = sf  + ((size_t)tok * 2 + kk) * DM;
    bf16*  db = sfb + ((size_t)tok * 2 + kk) * DM;
    for (int c = t; c < DM; c += 256) { float v = src[c]; d[c] = v; db[c] = f2bf(v); }
  }
}

// Collaborative attention: one block (256 thr) per token; L=2, nh=2, hd=384
__global__ __launch_bounds__(256) void collab_attn_kernel(
    const bf16* __restrict__ cqkv, bf16* __restrict__ aout, float* __restrict__ entbuf) {
  const int tok = blockIdx.x, t = threadIdx.x;
  __shared__ float qf[2 * TDQ];
  __shared__ float part[8 * 32];
  __shared__ float pr[8]; // [h][l][m]
  for (int i = t; i < 2 * TDQ; i += 256) {
    int l = i / TDQ, c = i % TDQ;
    qf[i] = bf2f(cqkv[((size_t)tok * 2 + l) * TDQ + c]);
  }
  __syncthreads();
  const int w = t >> 5, lane = t & 31;
  const int h = w >> 2, l = (w >> 1) & 1, m = w & 1;
  float s = 0.f;
  for (int j = lane; j < 384; j += 32)
    s += qf[l*TDQ + h*384 + j] * qf[m*TDQ + 768 + h*384 + j];
  part[w*32 + lane] = s;
  __syncthreads();
  if (t < 8) {
    float acc = 0.f;
    for (int j = 0; j < 32; ++j) acc += part[t*32 + j];
    part[t*32] = acc * 0.05103103630798287f; // 1/sqrt(384)
  }
  __syncthreads();
  if (t == 0) {
    float aw[4] = {0.f, 0.f, 0.f, 0.f};
    for (int hh = 0; hh < 2; ++hh)
      for (int ll = 0; ll < 2; ++ll) {
        float a = part[(hh*4 + ll*2 + 0) * 32];
        float b = part[(hh*4 + ll*2 + 1) * 32];
        float mx = fmaxf(a, b);
        float ea = expf(a - mx), eb = expf(b - mx), ss = ea + eb;
        pr[hh*4 + ll*2 + 0] = ea / ss;
        pr[hh*4 + ll*2 + 1] = eb / ss;
        aw[ll*2 + 0] += 0.5f * ea / ss;
        aw[ll*2 + 1] += 0.5f * eb / ss;
      }
    float ent = 0.f;
    for (int i = 0; i < 4; ++i) ent -= aw[i] * logf(aw[i] + 1e-9f);
    entbuf[tok] = ent; // sum over both rows; normalized in finalize
  }
  __syncthreads();
  for (int u = t; u < 1536; u += 256) {
    int hh = u / 768, rem = u % 768, ll = rem / 384, d = rem % 384;
    float o = pr[hh*4 + ll*2 + 0] * qf[0*TDQ + 1536 + hh*384 + d]
            + pr[hh*4 + ll*2 + 1] * qf[1*TDQ + 1536 + hh*384 + d];
    aout[((size_t)tok * 2 + ll) * DM + hh*384 + d] = f2bf(o);
  }
}

// Per-row RMS norm (collab): writes fp32 (for ffn residual) and bf16 (for ffn input)
__global__ __launch_bounds__(256) void rms_collab_kernel(
    const float* __restrict__ in, const float* __restrict__ w,
    float* __restrict__ outf, bf16* __restrict__ outb) {
  const int row = blockIdx.x, t = threadIdx.x;
  __shared__ float red[256];
  const float* r = in + (size_t)row * DM;
  float x0 = r[t], x1 = r[t + 256], x2 = r[t + 512];
  red[t] = x0*x0 + x1*x1 + x2*x2;
  __syncthreads();
  for (int s = 128; s > 0; s >>= 1) { if (t < s) red[t] += red[t + s]; __syncthreads(); }
  float inv = rsqrtf(red[0] / (float)DM + 1e-5f);
  float y0 = x0*inv*w[t], y1 = x1*inv*w[t+256], y2 = x2*inv*w[t+512];
  float* of = outf + (size_t)row * DM;
  bf16*  ob = outb + (size_t)row * DM;
  of[t] = y0; of[t+256] = y1; of[t+512] = y2;
  ob[t] = f2bf(y0); ob[t+256] = f2bf(y1); ob[t+512] = f2bf(y2);
}

// fused[tok][d] = sum_k refined[tok,k,d] * tkp[tok,k]  -> bf16
__global__ void weighted_sum_kernel(const float* __restrict__ refined,
                                    const float* __restrict__ tkp, bf16* __restrict__ fusedb) {
  int i = blockIdx.x * 256 + threadIdx.x;
  if (i >= NT * DM) return;
  int tok = i / DM, d = i % DM;
  float f = refined[((size_t)tok*2)     * DM + d] * tkp[tok*2]
          + refined[((size_t)tok*2 + 1) * DM + d] * tkp[tok*2 + 1];
  fusedb[(size_t)tok * DM + d] = f2bf(f);
}

// Deterministic aux-loss reduction (single block)
__global__ __launch_bounds__(256) void finalize_kernel(
    const float* __restrict__ probs8, const float* __restrict__ entbuf, float* __restrict__ aux_out) {
  __shared__ float red[256];
  const int t = threadIdx.x;
  float usage[NEXP];
  for (int e = 0; e < NEXP; ++e) {
    float s = 0.f;
    for (int tok = t; tok < NT; tok += 256) s += probs8[(size_t)tok * NEXP + e];
    red[t] = s; __syncthreads();
    for (int st = 128; st > 0; st >>= 1) { if (t < st) red[t] += red[t + st]; __syncthreads(); }
    usage[e] = red[0] / (float)NT;
    __syncthreads();
  }
  float s = 0.f;
  for (int tok = t; tok < NT; tok += 256) s += entbuf[tok];
  red[t] = s; __syncthreads();
  for (int st = 128; st > 0; st >>= 1) { if (t < st) red[t] += red[t + st]; __syncthreads(); }
  if (t == 0) {
    float div = red[0] / (float)(NT * 2);
    float mu = 0.f;
    for (int e = 0; e < NEXP; ++e) mu += usage[e];
    mu /= (float)NEXP;
    float var = 0.f;
    for (int e = 0; e < NEXP; ++e) { float d = usage[e] - mu; var += d * d; }
    var /= (float)NEXP;
    aux_out[0] = 0.1f * (0.01f * div + 0.01f * var);
  }
}

// ---------------------------------------------------------------------------
// Host launch
// ---------------------------------------------------------------------------
extern "C" void kernel_launch(void* const* d_in, const int* in_sizes, int n_in,
                              void* d_out, int out_size, void* d_ws, size_t ws_size,
                              hipStream_t stream) {
  const float* x       = (const float*)d_in[0];
  const float* w1      = (const float*)d_in[1];
  const float* w3      = (const float*)d_in[2];
  const float* w2      = (const float*)d_in[3];
  const float* r_in_w  = (const float*)d_in[4];
  const float* r_in_b  = (const float*)d_in[5];
  const float* r_out_w = (const float*)d_in[6];
  const float* r_out_b = (const float*)d_in[7];
  const float* r_norm  = (const float*)d_in[8];
  const float* gate_w  = (const float*)d_in[9];
  const float* c_in_w  = (const float*)d_in[10];
  const float* c_in_b  = (const float*)d_in[11];
  const float* c_out_w = (const float*)d_in[12];
  const float* c_out_b = (const float*)d_in[13];
  const float* c_norm  = (const float*)d_in[14];
  const float* ffn_w1  = (const float*)d_in[15];
  const float* ffn_w2  = (const float*)d_in[16];
  const float* o_w     = (const float*)d_in[17];
  float* out = (float*)d_out;
  const size_t OUT_AUX = (size_t)NT * DM;       // 3145728
  const size_t OUT_IDX = OUT_AUX + 1;

  char* ws = (char*)d_ws;
  size_t off = 0;
  auto alloc = [&](size_t bytes) -> void* {
    void* p = ws + off;
    off = (off + bytes + 255) & ~(size_t)255;
    return p;
  };

  bf16*  xb    = (bf16*) alloc((size_t)NT * DM * 2);
  bf16*  w1b   = (bf16*) alloc((size_t)NEXP * DM * HH * 2);
  bf16*  w3b   = (bf16*) alloc((size_t)NEXP * DM * HH * 2);
  bf16*  w2b   = (bf16*) alloc((size_t)NEXP * HH * DM * 2);
  bf16*  rinT  = (bf16*) alloc((size_t)DM * TDQ * 2);
  bf16*  routT = (bf16*) alloc((size_t)DM * DM * 2);
  bf16*  cinT  = (bf16*) alloc((size_t)DM * TDQ * 2);
  bf16*  coutT = (bf16*) alloc((size_t)DM * DM * 2);
  bf16*  f1T   = (bf16*) alloc((size_t)DM * DM * 2);
  bf16*  f2T   = (bf16*) alloc((size_t)DM * DM * 2);
  bf16*  owT   = (bf16*) alloc((size_t)DM * DM * 2);
  float* ef    = (float*)alloc((size_t)NE * DM * 4);
  bf16*  efb   = (bf16*) alloc((size_t)NE * DM * 2);
  float* ctxm  = (float*)alloc((size_t)NT * DM * 4);
  float* tkp   = (float*)alloc((size_t)NT * 2 * 4);
  float* probs8= (float*)alloc((size_t)NT * NEXP * 4);
  float* entbuf= (float*)alloc((size_t)NT * 4);
  bf16*  hbuf  = (bf16*) alloc((size_t)NT * HH * 2);
  float* sf    = (float*)alloc((size_t)NK * DM * 4);
  bf16*  sfb   = (bf16*) alloc((size_t)NK * DM * 2);

  // scratch region 1: router qkv (largest) reused by router tmp + whole collab phase
  char* scr1 = (char*)alloc((size_t)NE * TDQ * 2);
  bf16*  qkv     = (bf16*)scr1;                                   // NE x 2304 bf16
  float* tmp     = (float*)scr1;                                  // NE x 768 f32 (after qkv dead)
  bf16*  cqkv    = (bf16*) (scr1);                                // NK x 2304 bf16
  float* ao_tmp  = (float*)(scr1 + (size_t)NK * TDQ * 2);         // NK x 768 f32
  float* ao_f32  = (float*)(scr1 + (size_t)NK * TDQ * 2 + (size_t)NK * DM * 4);
  bf16*  aob     = (bf16*) (scr1 + (size_t)NK * TDQ * 2 + (size_t)NK * DM * 8);
  bf16*  ffn_mid = (bf16*) (scr1 + (size_t)NK * TDQ * 2 + (size_t)NK * DM * 10);
  float* refined = (float*)(scr1 + (size_t)NK * TDQ * 2 + (size_t)NK * DM * 12);

  // scratch region 2: attention outputs
  char* scr2 = (char*)alloc((size_t)NE * DM * 2);
  bf16* attn_out  = (bf16*)scr2;                                  // NE x 768
  bf16* attn2_out = (bf16*)scr2;                                  // NK x 768
  bf16* fusedb    = (bf16*)(scr2 + (size_t)NK * DM * 2);          // NT x 768

  auto g1 = [](int n) { return dim3((unsigned)((n + 255) / 256)); };

  // ---- 0) precision conversion / weight transposition ----
  cvt_f32_bf16_kernel<<<g1(NT*DM), 256, 0, stream>>>(x,  xb,  NT*DM);
  cvt_f32_bf16_kernel<<<g1(NEXP*DM*HH), 256, 0, stream>>>(w1, w1b, NEXP*DM*HH);
  cvt_f32_bf16_kernel<<<g1(NEXP*DM*HH), 256, 0, stream>>>(w3, w3b, NEXP*DM*HH);
  cvt_f32_bf16_kernel<<<g1(NEXP*HH*DM), 256, 0, stream>>>(w2, w2b, NEXP*HH*DM);
  transpose_cvt_kernel<<<g1(TDQ*DM), 256, 0, stream>>>(r_in_w,  rinT,  TDQ, DM);
  transpose_cvt_kernel<<<g1(DM*DM),  256, 0, stream>>>(r_out_w, routT, DM,  DM);
  transpose_cvt_kernel<<<g1(TDQ*DM), 256, 0, stream>>>(c_in_w,  cinT,  TDQ, DM);
  transpose_cvt_kernel<<<g1(DM*DM),  256, 0, stream>>>(c_out_w, coutT, DM,  DM);
  transpose_cvt_kernel<<<g1(DM*DM),  256, 0, stream>>>(ffn_w1,  f1T,   DM,  DM);
  transpose_cvt_kernel<<<g1(DM*DM),  256, 0, stream>>>(ffn_w2,  f2T,   DM,  DM);
  transpose_cvt_kernel<<<g1(DM*DM),  256, 0, stream>>>(o_w,     owT,   DM,  DM);

  // ---- 1) dense all-expert SwiGLU ----
  for (int e = 0; e < NEXP; ++e) {
    dual_gemm_silu_kernel<<<dim3(HH/GBN, NT/GBM), 256, 0, stream>>>(
        xb, DM, w1b + (size_t)e*DM*HH, w3b + (size_t)e*DM*HH, HH, DM, hbuf, HH);
    gemm_tpl<false,false,true,true,false><<<dim3(DM/GBN, NT/GBM), 256, 0, stream>>>(
        hbuf, HH, w2b + (size_t)e*HH*DM, DM, HH,
        nullptr, nullptr, 0,
        ef + (size_t)e*DM, NEXP*DM, efb + (size_t)e*DM, NEXP*DM);
  }

  // ---- 2) router self-attention over the 8 expert outputs per token ----
  gemm_tpl<true,false,false,true,false><<<dim3(TDQ/GBN, NE/GBM), 256, 0, stream>>>(
      efb, DM, rinT, TDQ, DM, r_in_b, nullptr, 0, nullptr, 0, qkv, TDQ);
  router_attn_kernel<<<dim3(NT * 12), 64, 0, stream>>>(qkv, attn_out);
  gemm_tpl<true,true,true,false,false><<<dim3(DM/GBN, NE/GBM), 256, 0, stream>>>(
      attn_out, DM, routT, DM, DM, r_out_b, ef, DM, tmp, DM, nullptr, 0);
  rms_mean_router_kernel<<<dim3(NT), 256, 0, stream>>>(tmp, r_norm, ctxm);

  // ---- 3) routing: logits, softmax, top-2, gather ----
  route_gather_kernel<<<dim3(NT), 256, 0, stream>>>(
      ctxm, gate_w, ef, sf, sfb, tkp, probs8, out + OUT_IDX);

  // ---- 4) collaborative attention + FFN over selected experts ----
  gemm_tpl<true,false,false,true,false><<<dim3(TDQ/GBN, NK/GBM), 256, 0, stream>>>(
      sfb, DM, cinT, TDQ, DM, c_in_b, nullptr, 0, nullptr, 0, cqkv, TDQ);
  collab_attn_kernel<<<dim3(NT), 256, 0, stream>>>(cqkv, attn2_out, entbuf);
  gemm_tpl<true,true,true,false,false><<<dim3(DM/GBN, NK/GBM), 256, 0, stream>>>(
      attn2_out, DM, coutT, DM, DM, c_out_b, sf, DM, ao_tmp, DM, nullptr, 0);
  rms_collab_kernel<<<dim3(NK), 256, 0, stream>>>(ao_tmp, c_norm, ao_f32, aob);
  gemm_tpl<false,false,false,true,true><<<dim3(DM/GBN, NK/GBM), 256, 0, stream>>>(
      aob, DM, f1T, DM, DM, nullptr, nullptr, 0, nullptr, 0, ffn_mid, DM);
  gemm_tpl<false,true,true,false,false><<<dim3(DM/GBN, NK/GBM), 256, 0, stream>>>(
      ffn_mid, DM, f2T, DM, DM, nullptr, ao_f32, DM, refined, DM, nullptr, 0);

  // ---- 5) weighted fusion + output projection + aux ----
  weighted_sum_kernel<<<g1(NT*DM), 256, 0, stream>>>(refined, tkp, fusedb);
  gemm_tpl<false,false,true,false,false><<<dim3(DM/GBN, NT/GBM), 256, 0, stream>>>(
      fusedb, DM, owT, DM, DM, nullptr, nullptr, 0, out, DM, nullptr, 0);
  finalize_kernel<<<dim3(1), 256, 0, stream>>>(probs8, entbuf, out + OUT_AUX);
}
